// Attention_53944789238363
// MI455X (gfx1250) — matrix operand order repaired and checked by reading, not compiled, not run on hardware
//
#include <hip/hip_runtime.h>
#include <hip/hip_bf16.h>
#include <math.h>

// ---------------------------------------------------------------------------
// Problem constants (from reference)
// ---------------------------------------------------------------------------
#define N_NODES   50000
#define N_EDGES   800000
#define Q_DIM     256
#define PAIR_DIM  64
#define NUM_HEADS 8
#define HEAD_DIM  16
#define TOTAL_DIM 128   // NUM_HEADS * HEAD_DIM

typedef __attribute__((ext_vector_type(16))) __bf16 v16bf;
typedef __attribute__((ext_vector_type(8)))  float  v8f;

__device__ __forceinline__ __bf16 f2bf(float f) { return (__bf16)f; }

// CAS-based float atomic max (portable; logits can be negative)
__device__ __forceinline__ void atomicMaxF32(float* addr, float val) {
    unsigned int* ua = (unsigned int*)addr;
    unsigned int old = *ua;
    while (true) {
        float cur = __uint_as_float(old);
        if (cur >= val) break;
        unsigned int assumed = old;
        old = atomicCAS(ua, assumed, __float_as_uint(val));
        if (old == assumed) break;
    }
}

// ---------------------------------------------------------------------------
// Kernel 0: initialize workspace (m = -inf, denom = 0, o = 0)
// ---------------------------------------------------------------------------
__global__ void init_state_kernel(float* __restrict__ m, float* __restrict__ denom,
                                  float* __restrict__ o, int nm, int no) {
    int i = blockIdx.x * blockDim.x + threadIdx.x;
    if (i < nm) { m[i] = -__builtin_inff(); denom[i] = 0.0f; }
    if (i < no) { o[i] = 0.0f; }
}

// ---------------------------------------------------------------------------
// GEMM: Y[M,N] = scale * (X[M,K] @ W[N,K]^T) (+ bias[N])
// bf16 WMMA 16x16x32, fp32 accumulate.
//
// Block = 256 threads = 8 waves. The block stages an NT*16-row strip of W
// (rows [nBase, nBase+NT*16), all K columns) into LDS as bf16 once; each wave
// owns one 16-row m-tile and computes all NT n-tiles of the strip, so each
// A fragment (global load + cvt) is reused by NT WMMAs, and B fragments come
// from LDS (ds_load_b128), never re-converted.
//
// Fragment layouts (ISA 7.12.2, wave32):
//   A (16x32 bf16): lanes 0-15: M=lane,   elems[0..7]=K k+0..7, [8..15]=k+16..23
//                   lanes 16-31: M=lane-16, K k+8..15 / k+24..31
//   B (32x16 bf16): lanes 0-15: N=lane,   elems[0..7]=K k+0..7, [8..15]=k+16..23
//                   lanes 16-31: N=lane-16, K k+8..15 / k+24..31
//   D (16x16 f32):  VGPR r: lanes 0-15 -> (m0+r, n0+lane);
//                   lanes 16-31 -> (m0+8+r, n0+lane-16)
// Requires M%16==0, K%32==0, N%(NT*16)==0 (true here).
// ---------------------------------------------------------------------------
template <int N, int K, int NT, bool HAS_BIAS>
__global__ void gemm_bf16_wmma_kernel(const float* __restrict__ X,
                                      const float* __restrict__ W,
                                      const float* __restrict__ bias,
                                      float* __restrict__ Y,
                                      int M, float scale) {
    __shared__ __bf16 sW[NT * 16 * K];   // proj: 64KB, final strip: 32KB

    const int lane = threadIdx.x & 31;
    const int wave = threadIdx.x >> 5;
    const int nBase = blockIdx.y * (NT * 16);

    // Cooperative stage: W strip -> LDS (bf16), float4-vectorized
    {
        const float* __restrict__ wsrc = W + (size_t)nBase * K;
        const int total4 = (NT * 16 * K) / 4;
        for (int i4 = threadIdx.x; i4 < total4; i4 += blockDim.x) {
            const float4 wv = ((const float4*)wsrc)[i4];
            sW[i4 * 4 + 0] = f2bf(wv.x);
            sW[i4 * 4 + 1] = f2bf(wv.y);
            sW[i4 * 4 + 2] = f2bf(wv.z);
            sW[i4 * 4 + 3] = f2bf(wv.w);
        }
    }
    __syncthreads();

    const int mTile = blockIdx.x * (blockDim.x >> 5) + wave;
    const int m0 = mTile * 16;
    if (m0 >= M) return;   // wave-uniform exit (after barrier)

    const int half = lane >> 4;   // 0: lanes 0-15, 1: lanes 16-31
    const int l16  = lane & 15;
    const float* __restrict__ aBase = X + (size_t)(m0 + l16) * K;

    v8f acc[NT];
#pragma unroll
    for (int t = 0; t < NT; ++t) acc[t] = (v8f){};

    for (int k = 0; k < K; k += 32) {
        if (k + 32 < K) __builtin_prefetch(aBase + k + 32, 0, 1);  // global_prefetch_b8
        // One A fragment per k-step, reused by NT WMMAs
        const float* ap = aBase + k + half * 8;
        v16bf a;
#pragma unroll
        for (int i = 0; i < 8; ++i) {
            a[i]     = f2bf(ap[i]);
            a[i + 8] = f2bf(ap[16 + i]);
        }
#pragma unroll
        for (int t = 0; t < NT; ++t) {
            const __bf16* bp = sW + (size_t)(t * 16 + l16) * K + k + half * 8;
            v16bf b;
#pragma unroll
            for (int i = 0; i < 8; ++i) { b[i] = bp[i]; b[i + 8] = bp[16 + i]; }   // 2x16B -> ds_load_b128 x2
            acc[t] = __builtin_amdgcn_wmma_f32_16x16x32_bf16(
                /*neg_a=*/false, a, /*neg_b=*/false, b,
                /*c_mod=*/(short)0, acc[t], /*reuse_a=*/false, /*reuse_b=*/false);
        }
    }

#pragma unroll
    for (int t = 0; t < NT; ++t) {
        const int col = nBase + t * 16 + l16;
        float badd = 0.0f;
        if (HAS_BIAS) badd = bias[col];
#pragma unroll
        for (int r = 0; r < 8; ++r) {
            const int row = m0 + half * 8 + r;
            Y[(size_t)row * N + col] = acc[t][r] * scale + badd;
        }
    }
}

// ---------------------------------------------------------------------------
// Kernel 2: attn[e,h] = bb[h] + edges[e,:] . Wb[h,:]   (streams 205MB edges)
// ---------------------------------------------------------------------------
__global__ void edge_bias_kernel(const float* __restrict__ edges,
                                 const float* __restrict__ Wb,
                                 const float* __restrict__ bb,
                                 float* __restrict__ attn, int E) {
    __shared__ float sWb[NUM_HEADS * PAIR_DIM];
    __shared__ float sbb[NUM_HEADS];
    for (int i = threadIdx.x; i < NUM_HEADS * PAIR_DIM; i += blockDim.x) sWb[i] = Wb[i];
    if (threadIdx.x < NUM_HEADS) sbb[threadIdx.x] = bb[threadIdx.x];
    __syncthreads();

    const int e = blockIdx.x * blockDim.x + threadIdx.x;
    if (e >= E) return;

    float acc[NUM_HEADS];
#pragma unroll
    for (int h = 0; h < NUM_HEADS; ++h) acc[h] = sbb[h];

    const float4* __restrict__ er = (const float4*)(edges + (size_t)e * PAIR_DIM);
#pragma unroll
    for (int j4 = 0; j4 < PAIR_DIM / 4; ++j4) {
        const float4 ev = er[j4];
#pragma unroll
        for (int h = 0; h < NUM_HEADS; ++h) {
            const float* wr = sWb + h * PAIR_DIM + j4 * 4;
            acc[h] = fmaf(ev.x, wr[0], acc[h]);
            acc[h] = fmaf(ev.y, wr[1], acc[h]);
            acc[h] = fmaf(ev.z, wr[2], acc[h]);
            acc[h] = fmaf(ev.w, wr[3], acc[h]);
        }
    }
#pragma unroll
    for (int h = 0; h < NUM_HEADS; ++h) attn[(size_t)e * NUM_HEADS + h] = acc[h];
}

// ---------------------------------------------------------------------------
// Kernel 3: attn[e,h] += qh[src,h,:].kh[dst,h,:]; segment max into m[src,h]
// Gathers hit L2 (qh+kh = 51MB << 192MB L2).
// ---------------------------------------------------------------------------
__global__ void edge_logits_kernel(const float* __restrict__ qh,
                                   const float* __restrict__ kh,
                                   const int* __restrict__ ei,
                                   float* __restrict__ attn,
                                   float* __restrict__ m, int E) {
    const int gid = blockIdx.x * blockDim.x + threadIdx.x;
    if (gid >= E * NUM_HEADS) return;
    const int e = gid >> 3, h = gid & 7;
    const int src = ei[e * 2 + 0];
    const int dst = ei[e * 2 + 1];
    const float4* qp = (const float4*)(qh + (size_t)src * TOTAL_DIM + h * HEAD_DIM);
    const float4* kp = (const float4*)(kh + (size_t)dst * TOTAL_DIM + h * HEAD_DIM);
    float d = 0.0f;
#pragma unroll
    for (int i = 0; i < HEAD_DIM / 4; ++i) {
        const float4 a = qp[i], b = kp[i];
        d = fmaf(a.x, b.x, d); d = fmaf(a.y, b.y, d);
        d = fmaf(a.z, b.z, d); d = fmaf(a.w, b.w, d);
    }
    const float logit = attn[gid] + d;
    attn[gid] = logit;
    atomicMaxF32(m + (size_t)src * NUM_HEADS + h, logit);
}

// ---------------------------------------------------------------------------
// Kernel 4: p = exp(attn - m[src]); segment sum into denom[src,h]
// ---------------------------------------------------------------------------
__global__ void edge_exp_kernel(const int* __restrict__ ei,
                                const float* __restrict__ m,
                                float* __restrict__ attn,
                                float* __restrict__ denom, int E) {
    const int gid = blockIdx.x * blockDim.x + threadIdx.x;
    if (gid >= E * NUM_HEADS) return;
    const int e = gid >> 3, h = gid & 7;
    const int src = ei[e * 2 + 0];
    const float p = expf(attn[gid] - m[(size_t)src * NUM_HEADS + h]);
    attn[gid] = p;
    atomicAdd(denom + (size_t)src * NUM_HEADS + h, p);
}

// ---------------------------------------------------------------------------
// Kernel 5: o[src,h,:] += (p/denom[src,h]) * vh[dst,h,:]
// ---------------------------------------------------------------------------
__global__ void edge_aggregate_kernel(const int* __restrict__ ei,
                                      const float* __restrict__ attn,
                                      const float* __restrict__ denom,
                                      const float* __restrict__ vh,
                                      float* __restrict__ o, int E) {
    const int gid = blockIdx.x * blockDim.x + threadIdx.x;
    if (gid >= E * NUM_HEADS) return;
    const int e = gid >> 3, h = gid & 7;
    const int src = ei[e * 2 + 0];
    const int dst = ei[e * 2 + 1];
    const float p = attn[gid] / denom[(size_t)src * NUM_HEADS + h];
    const float4* vp = (const float4*)(vh + (size_t)dst * TOTAL_DIM + h * HEAD_DIM);
    float* op = o + (size_t)src * TOTAL_DIM + h * HEAD_DIM;
#pragma unroll
    for (int i = 0; i < HEAD_DIM / 4; ++i) {
        const float4 vv = vp[i];
        atomicAdd(op + i * 4 + 0, p * vv.x);
        atomicAdd(op + i * 4 + 1, p * vv.y);
        atomicAdd(op + i * 4 + 2, p * vv.z);
        atomicAdd(op + i * 4 + 3, p * vv.w);
    }
}

// ---------------------------------------------------------------------------
// Host-side launch
// ---------------------------------------------------------------------------
extern "C" void kernel_launch(void* const* d_in, const int* in_sizes, int n_in,
                              void* d_out, int out_size, void* d_ws, size_t ws_size,
                              hipStream_t stream) {
    (void)in_sizes; (void)n_in; (void)out_size; (void)ws_size;

    const float* q     = (const float*)d_in[0];
    const float* k     = (const float*)d_in[1];
    const float* v     = (const float*)d_in[2];
    const float* edges = (const float*)d_in[3];
    const int*   ei    = (const int*)  d_in[4];
    const float* Wq    = (const float*)d_in[5];
    const float* Wk    = (const float*)d_in[6];
    const float* Wv    = (const float*)d_in[7];
    const float* Wb    = (const float*)d_in[8];
    const float* bb    = (const float*)d_in[9];
    const float* Wo    = (const float*)d_in[10];
    const float* bo    = (const float*)d_in[11];
    float* out = (float*)d_out;

    // Workspace layout (floats):
    float* ws   = (float*)d_ws;
    float* qh   = ws;                                   // N*128
    float* kh   = qh + (size_t)N_NODES * TOTAL_DIM;     // N*128
    float* vh   = kh + (size_t)N_NODES * TOTAL_DIM;     // N*128
    float* o    = vh + (size_t)N_NODES * TOTAL_DIM;     // N*128
    float* mbuf = o  + (size_t)N_NODES * TOTAL_DIM;     // N*8
    float* dnm  = mbuf + (size_t)N_NODES * NUM_HEADS;   // N*8
    float* attn = dnm  + (size_t)N_NODES * NUM_HEADS;   // E*8

    const float norm = 0.25f;   // HEAD_DIM^-0.5
    const int mTiles = N_NODES / 16;            // 3125
    const int mBlocks = (mTiles + 7) / 8;       // 8 waves per block -> 391

    // 0) init
    {
        const int no = N_NODES * TOTAL_DIM;
        const int nm = N_NODES * NUM_HEADS;
        init_state_kernel<<<(no + 255) / 256, 256, 0, stream>>>(mbuf, dnm, o, nm, no);
    }

    // 1) projections: qh, kh, vh  (M=50000, N=128, K=256; whole W in LDS)
    {
        dim3 blk(256, 1, 1);
        dim3 grd(mBlocks, 1, 1);   // grid.y = N/(NT*16) = 1
        gemm_bf16_wmma_kernel<TOTAL_DIM, Q_DIM, 8, false>
            <<<grd, blk, 0, stream>>>(q, Wq, nullptr, qh, N_NODES, norm);
        gemm_bf16_wmma_kernel<TOTAL_DIM, Q_DIM, 8, false>
            <<<grd, blk, 0, stream>>>(k, Wk, nullptr, kh, N_NODES, 1.0f);
        gemm_bf16_wmma_kernel<TOTAL_DIM, Q_DIM, 8, false>
            <<<grd, blk, 0, stream>>>(v, Wv, nullptr, vh, N_NODES, 1.0f);
    }

    // 2) edge bias
    edge_bias_kernel<<<(N_EDGES + 255) / 256, 256, 0, stream>>>(edges, Wb, bb, attn, N_EDGES);

    // 3) logits + segment max
    {
        const int tot = N_EDGES * NUM_HEADS;
        edge_logits_kernel<<<(tot + 255) / 256, 256, 0, stream>>>(qh, kh, ei, attn, mbuf, N_EDGES);
    }

    // 4) exp + segment sum
    {
        const int tot = N_EDGES * NUM_HEADS;
        edge_exp_kernel<<<(tot + 255) / 256, 256, 0, stream>>>(ei, mbuf, attn, dnm, N_EDGES);
    }

    // 5) normalize + aggregate
    {
        const int tot = N_EDGES * NUM_HEADS;
        edge_aggregate_kernel<<<(tot + 255) / 256, 256, 0, stream>>>(ei, attn, dnm, vh, o, N_EDGES);
    }

    // 6) output projection: out = o @ Wo.T + bo  (M=50000, N=256, K=128)
    {
        dim3 blk(256, 1, 1);
        dim3 grd(mBlocks, 2, 1);   // grid.y = 256/(8*16) = 2 column strips
        gemm_bf16_wmma_kernel<Q_DIM, TOTAL_DIM, 8, true>
            <<<grd, blk, 0, stream>>>(o, Wo, bo, out, N_NODES, 1.0f);
    }
}
